// GATNet_1_9732395892847
// MI455X (gfx1250) — matrix operand order repaired and checked by reading, not compiled, not run on hardware
//
#include <hip/hip_runtime.h>

// ---------------- constants (from reference) ----------------
#define N_NODES 40000
#define N_EDGES 160000
#define N_FEATS 74
#define KF      96          // padded feature K for WMMA (74 -> 96 = 3*32)
#define HH      3
#define DD      1024
#define HD      3072        // H*D
#define NPG     50
#define NG      800

typedef __bf16 bf16_t;
typedef __attribute__((ext_vector_type(16))) __bf16 bf16x16;
typedef __attribute__((ext_vector_type(8)))  __bf16 bf16x8;
typedef __attribute__((ext_vector_type(2)))  __bf16 bf16x2;
typedef __attribute__((ext_vector_type(8)))  float  f32x8;
typedef __attribute__((ext_vector_type(4)))  float  f32x4;

// ---------------- helpers ----------------
__device__ __forceinline__ void atomicAddF32(float* p, float v) {
  unsafeAtomicAdd(p, v);   // lowers to global_atomic_add_f32 on gfx1250
}
// order-preserving float -> uint encoding for atomicMax-based segment max
__device__ __forceinline__ unsigned fenc(float f) {
  unsigned u = __float_as_uint(f);
  return (u & 0x80000000u) ? ~u : (u | 0x80000000u);
}
__device__ __forceinline__ float fdec(unsigned k) {
  unsigned u = (k & 0x80000000u) ? (k ^ 0x80000000u) : ~k;
  return __uint_as_float(u);
}

// ---------------- prep kernels ----------------
__global__ void k_prep_feats(const float* __restrict__ feats, bf16_t* __restrict__ featsB) {
  int t = blockIdx.x * 256 + threadIdx.x;
  if (t >= N_NODES * KF) return;
  int n = t / KF, k = t % KF;
  featsB[t] = (k < N_FEATS) ? (bf16_t)feats[(size_t)n * N_FEATS + k] : (bf16_t)0.0f;
}
__global__ void k_prep_WT(const float* __restrict__ W, bf16_t* __restrict__ WT) {
  int t = blockIdx.x * 256 + threadIdx.x;       // WT[j][k], j<HD, k<KF
  if (t >= HD * KF) return;
  int j = t / KF, k = t % KF;
  WT[t] = (k < N_FEATS) ? (bf16_t)W[(size_t)k * HD + j] : (bf16_t)0.0f;
}
__global__ void k_prep_W1T(const float* __restrict__ W1, bf16_t* __restrict__ W1T) {
  int t = blockIdx.x * 256 + threadIdx.x;       // W1T[j][k], j<DD, k<HD
  if (t >= DD * HD) return;
  int j = t / HD, k = t % HD;
  W1T[t] = (bf16_t)W1[(size_t)k * DD + j];
}
// biasW1[j] = b1[j] + sum_k bias_gat[k]*W1[k][j]   (folds (out+bias)@W1 bias term)
__global__ void k_biasW1(const float* __restrict__ bias_gat, const float* __restrict__ W1,
                         const float* __restrict__ b1, float* __restrict__ biasW1) {
  int j = blockIdx.x * 256 + threadIdx.x;
  if (j >= DD) return;
  float s = b1[j];
  for (int k = 0; k < HD; ++k) s += bias_gat[k] * W1[(size_t)k * DD + j];
  biasW1[j] = s;
}

// ---------------- WMMA GEMM: C[M,N] = A[M,K] @ BT[N,K]^T (+bias[N]) ----------------
// block = 256 threads = 8 waves; block tile 128(M) x 128(N); wave tile 32x64.
template<int KTOT, bool A_F32, bool OUT_BF16, bool HAS_BIAS>
__global__ __launch_bounds__(256)
void gemm_wmma(const void* __restrict__ Aptr, const bf16_t* __restrict__ BT,
               const float* __restrict__ biasN, void* __restrict__ Cptr,
               int M, int Nn) {
  const int lane = threadIdx.x & 31;
  const int wave = threadIdx.x >> 5;
  const int wm = wave & 3, wn = wave >> 2;
  const int mBase = blockIdx.x * 128 + wm * 32;
  const int nBase = blockIdx.y * 128 + wn * 64;
  const int mrow = lane & 15, half = lane >> 4;
  const int ncol = lane & 15;

  // hoisted fragment base pointers (k-invariant)
  const char* aRow[2];
#pragma unroll
  for (int t = 0; t < 2; ++t) {
    int row = mBase + t * 16 + mrow;
    if (row >= M) row = M - 1;                 // clamp: junk rows never stored
    if constexpr (A_F32)
      aRow[t] = (const char*)((const float*)Aptr + (size_t)row * KTOT);
    else
      aRow[t] = (const char*)((const bf16_t*)Aptr + (size_t)row * KTOT);
  }
  const bf16_t* bPtr[4];
#pragma unroll
  for (int t = 0; t < 4; ++t)
    bPtr[t] = BT + (size_t)(nBase + t * 16 + ncol) * KTOT + half * 8;

  f32x8 acc[2][4];
#pragma unroll
  for (int i = 0; i < 2; ++i)
#pragma unroll
    for (int j = 0; j < 4; ++j) acc[i][j] = (f32x8)0.0f;

  for (int k0 = 0; k0 < KTOT; k0 += 32) {
    // A fragments: lane holds row, K = {half*8..+7} U {16+half*8..+7}
    bf16x16 a[2];
#pragma unroll
    for (int t = 0; t < 2; ++t) {
      if constexpr (A_F32) {
        const float* p = (const float*)aRow[t] + k0;
        // prefetch next A tile while this one feeds the WMMAs
        if (k0 + 32 < KTOT) __builtin_prefetch((const void*)(p + 32), 0, 0);
        f32x4 c00 = *(const f32x4*)(p + half * 8);
        f32x4 c01 = *(const f32x4*)(p + half * 8 + 4);
        f32x4 c10 = *(const f32x4*)(p + 16 + half * 8);
        f32x4 c11 = *(const f32x4*)(p + 16 + half * 8 + 4);
#pragma unroll
        for (int i = 0; i < 4; ++i) {
          a[t][i]      = (bf16_t)c00[i];
          a[t][4 + i]  = (bf16_t)c01[i];
          a[t][8 + i]  = (bf16_t)c10[i];
          a[t][12 + i] = (bf16_t)c11[i];
        }
      } else {
        const bf16_t* p = (const bf16_t*)aRow[t] + k0;
        bf16x8 c0 = *(const bf16x8*)(p + half * 8);
        bf16x8 c1 = *(const bf16x8*)(p + 16 + half * 8);
#pragma unroll
        for (int i = 0; i < 8; ++i) { a[t][i] = c0[i]; a[t][8 + i] = c1[i]; }
      }
    }
    // B fragments from column-major BT[n][k]: lane holds col, K = {half*8..+7} U {16+half*8..+7}
    bf16x16 b[4];
#pragma unroll
    for (int t = 0; t < 4; ++t) {
      const bf16_t* p = bPtr[t] + k0;
      bf16x8 c0 = *(const bf16x8*)(p);
      bf16x8 c1 = *(const bf16x8*)(p + 16);
#pragma unroll
      for (int i = 0; i < 8; ++i) { b[t][i] = c0[i]; b[t][8 + i] = c1[i]; }
    }
#pragma unroll
    for (int i = 0; i < 2; ++i)
#pragma unroll
      for (int j = 0; j < 4; ++j)
        acc[i][j] = __builtin_amdgcn_wmma_f32_16x16x32_bf16(
            false, a[i], false, b[j], (short)0, acc[i][j], false, false);
  }

  // --- epilogue: C/D layout = row (mBase + i*16 + half*8 + r), col (nBase + j*16 + ncol)
  const bool interior = (mBase + 32) <= M;     // true for 312 of 313 M-blocks
  if (interior) {
#pragma unroll
    for (int i = 0; i < 2; ++i)
#pragma unroll
      for (int j = 0; j < 4; ++j) {
        int col = nBase + j * 16 + ncol;
        float bv = HAS_BIAS ? biasN[col] : 0.0f;
#pragma unroll
        for (int r = 0; r < 8; ++r) {
          int row = mBase + i * 16 + half * 8 + r;
          float v = acc[i][j][r] + bv;
          if constexpr (OUT_BF16)
            ((bf16_t*)Cptr)[(size_t)row * Nn + col] = (bf16_t)v;
          else
            ((float*)Cptr)[(size_t)row * Nn + col] = v;
        }
      }
  } else {
#pragma unroll
    for (int i = 0; i < 2; ++i)
#pragma unroll
      for (int j = 0; j < 4; ++j) {
        int col = nBase + j * 16 + ncol;
        float bv = HAS_BIAS ? biasN[col] : 0.0f;
#pragma unroll
        for (int r = 0; r < 8; ++r) {
          int row = mBase + i * 16 + half * 8 + r;
          if (row < M) {
            float v = acc[i][j][r] + bv;
            if constexpr (OUT_BF16)
              ((bf16_t*)Cptr)[(size_t)row * Nn + col] = (bf16_t)v;
            else
              ((float*)Cptr)[(size_t)row * Nn + col] = v;
          }
        }
      }
  }
}

// ---------------- el/er: one wave per (node, head) ----------------
__global__ void k_el_er(const bf16_t* __restrict__ hB, const float* __restrict__ attn_l,
                        const float* __restrict__ attn_r, float* __restrict__ el,
                        float* __restrict__ er) {
  int gw = blockIdx.x * 8 + (threadIdx.x >> 5);
  int lane = threadIdx.x & 31;
  if (gw >= N_NODES * HH) return;
  int n = gw / HH, hh = gw % HH;
  const bf16_t* hp = hB + (size_t)n * HD + hh * DD;
  const float* al = attn_l + hh * DD;
  const float* ar = attn_r + hh * DD;
  float sl = 0.f, sr = 0.f;
#pragma unroll
  for (int i = 0; i < 32; ++i) {
    int d = i * 32 + lane;
    float hv = (float)hp[d];
    sl += hv * al[d];
    sr += hv * ar[d];
  }
#pragma unroll
  for (int o = 16; o > 0; o >>= 1) {
    sl += __shfl_xor(sl, o, 32);
    sr += __shfl_xor(sr, o, 32);
  }
  if (lane == 0) { el[gw] = sl; er[gw] = sr; }
}

// ---------------- edge softmax ----------------
__global__ void k_edge_max(const float* __restrict__ el, const float* __restrict__ er,
                           const int* __restrict__ src, const int* __restrict__ dst,
                           float* __restrict__ eBuf, unsigned* __restrict__ mkey) {
  int t = blockIdx.x * 256 + threadIdx.x;
  if (t >= N_EDGES * HH) return;
  int e = t / HH, hh = t % HH;
  float v = el[src[e] * HH + hh] + er[dst[e] * HH + hh];
  v = v > 0.f ? v : 0.2f * v;                 // leaky_relu(0.2)
  eBuf[t] = v;
  atomicMax(&mkey[dst[e] * HH + hh], fenc(v));
}
__global__ void k_edge_exp_sum(const int* __restrict__ dst, const unsigned* __restrict__ mkey,
                               float* __restrict__ eBuf, float* __restrict__ sB) {
  int t = blockIdx.x * 256 + threadIdx.x;
  if (t >= N_EDGES * HH) return;
  int e = t / HH, hh = t % HH;
  int di = dst[e] * HH + hh;
  float ex = __expf(eBuf[t] - fdec(mkey[di]));
  eBuf[t] = ex;
  atomicAddF32(&sB[di], ex);
}

// ---------------- aggregation: one wave per (edge, head) ----------------
__global__ void k_aggregate(const int* __restrict__ src, const int* __restrict__ dst,
                            const float* __restrict__ eBuf, const float* __restrict__ sB,
                            const bf16_t* __restrict__ hB, float* __restrict__ outAgg) {
  int gw = blockIdx.x * 8 + (threadIdx.x >> 5);
  int lane = threadIdx.x & 31;
  if (gw >= N_EDGES * HH) return;
  int e = gw / HH, hh = gw % HH;
  int d0 = dst[e], s0 = src[e];
  float alpha = eBuf[gw] / sB[d0 * HH + hh];
  const bf16_t* hp = hB + (size_t)s0 * HD + hh * DD;
  float* op = outAgg + (size_t)d0 * HD + hh * DD;
#pragma unroll
  for (int i = 0; i < 16; ++i) {
    int idx = i * 64 + lane * 2;
    bf16x2 hv = *(const bf16x2*)(hp + idx);
    atomicAddF32(op + idx,     (float)hv[0] * alpha);
    atomicAddF32(op + idx + 1, (float)hv[1] * alpha);
  }
}

// ---------------- graph mean-pool + final linear ----------------
__global__ void k_pool(const float* __restrict__ x, const float* __restrict__ W2,
                       const float* __restrict__ b2, float* __restrict__ y) {
  __shared__ float red[256];
  int g = blockIdx.x, t = threadIdx.x;
  float p = 0.f;
  for (int n = g * NPG; n < (g + 1) * NPG; ++n) {
    const float* row = x + (size_t)n * DD;
    for (int d = t; d < DD; d += 256) p += row[d] * W2[d];
  }
  red[t] = p;
  __syncthreads();
  for (int s = 128; s > 0; s >>= 1) {
    if (t < s) red[t] += red[t + s];
    __syncthreads();
  }
  if (t == 0) y[g] = red[0] / (float)NPG + b2[0];
}

// ---------------- launcher ----------------
extern "C" void kernel_launch(void* const* d_in, const int* in_sizes, int n_in,
                              void* d_out, int out_size, void* d_ws, size_t ws_size,
                              hipStream_t stream) {
  const float* feats    = (const float*)d_in[0];
  const float* W        = (const float*)d_in[1];
  const float* attn_l   = (const float*)d_in[2];
  const float* attn_r   = (const float*)d_in[3];
  const float* bias_gat = (const float*)d_in[4];
  const float* W1       = (const float*)d_in[5];
  const float* b1       = (const float*)d_in[6];
  const float* W2       = (const float*)d_in[7];
  const float* b2       = (const float*)d_in[8];
  const int*   src      = (const int*)d_in[9];
  const int*   dst      = (const int*)d_in[10];
  float* y = (float*)d_out;

  // workspace carve (256B aligned)
  char* w = (char*)d_ws;
  auto carve = [&](size_t bytes) { void* p = (void*)w; w += (bytes + 255) & ~(size_t)255; return p; };
  bf16_t*   hB     = (bf16_t*)  carve((size_t)N_NODES * HD * 2);   // 245.8 MB
  float*    outAgg = (float*)   carve((size_t)N_NODES * HD * 4);   // 491.5 MB
  float*    xbuf   = (float*)   carve((size_t)N_NODES * DD * 4);   // 163.8 MB
  bf16_t*   featsB = (bf16_t*)  carve((size_t)N_NODES * KF * 2);
  bf16_t*   WT     = (bf16_t*)  carve((size_t)HD * KF * 2);
  bf16_t*   W1T    = (bf16_t*)  carve((size_t)DD * HD * 2);
  float*    el     = (float*)   carve((size_t)N_NODES * HH * 4);
  float*    er     = (float*)   carve((size_t)N_NODES * HH * 4);
  float*    eBuf   = (float*)   carve((size_t)N_EDGES * HH * 4);
  unsigned* mkey   = (unsigned*)carve((size_t)N_NODES * HH * 4);
  float*    sB     = (float*)   carve((size_t)N_NODES * HH * 4);
  float*    biasW1 = (float*)   carve((size_t)DD * 4);

  // zero accumulators (graph-capture-safe)
  hipMemsetAsync(outAgg, 0, (size_t)N_NODES * HD * 4, stream);
  hipMemsetAsync(mkey,   0, (size_t)N_NODES * HH * 4, stream);
  hipMemsetAsync(sB,     0, (size_t)N_NODES * HH * 4, stream);

  // prep
  k_prep_feats<<<(N_NODES * KF + 255) / 256, 256, 0, stream>>>(feats, featsB);
  k_prep_WT  <<<(HD * KF + 255) / 256, 256, 0, stream>>>(W, WT);
  k_prep_W1T <<<(DD * HD + 255) / 256, 256, 0, stream>>>(W1, W1T);
  k_biasW1   <<<(DD + 255) / 256, 256, 0, stream>>>(bias_gat, W1, b1, biasW1);

  // GEMM1: h = feats @ W  (bf16 out)
  gemm_wmma<KF, false, true, false>
      <<<dim3((N_NODES + 127) / 128, HD / 128), 256, 0, stream>>>(featsB, WT, nullptr, hB, N_NODES, HD);

  // attention scalars
  k_el_er<<<(N_NODES * HH + 7) / 8, 256, 0, stream>>>(hB, attn_l, attn_r, el, er);

  // edge softmax
  k_edge_max    <<<(N_EDGES * HH + 255) / 256, 256, 0, stream>>>(el, er, src, dst, eBuf, mkey);
  k_edge_exp_sum<<<(N_EDGES * HH + 255) / 256, 256, 0, stream>>>(dst, mkey, eBuf, sB);

  // message aggregation
  k_aggregate<<<(N_EDGES * HH + 7) / 8, 256, 0, stream>>>(src, dst, eBuf, sB, hB, outAgg);

  // GEMM2: x = out @ W1 + (bias_gat@W1 + b1)   (f32 out, fused bias)
  gemm_wmma<HD, true, false, true>
      <<<dim3((N_NODES + 127) / 128, DD / 128), 256, 0, stream>>>(outAgg, W1T, biasW1, xbuf, N_NODES, DD);

  // mean pool per graph (contiguous 50-node blocks) + final linear
  k_pool<<<NG, 256, 0, stream>>>(xbuf, W2, b2, y);
}
